// Block_19576460935839
// MI455X (gfx1250) — matrix operand order repaired and checked by reading, not compiled, not run on hardware
//
#include <hip/hip_runtime.h>
#include <hip/hip_bf16.h>
#include <math.h>

// ---------------------------------------------------------------------------
// Transformer block for MI455X (gfx1250, wave32, WMMA).
// All GEMMs run through a single bf16 WMMA kernel (v_wmma_f32_16x16x32_bf16),
// f32 accumulation. Attention materializes the 64MB score matrix (fits in the
// 192MB L2) instead of flash-attention (head dim 1024 makes register-resident
// O accumulators impossible).
//
// GEMM main loop: double-buffered LDS, one barrier per BK=32 step, next tile
// fetched with GLOBAL_LOAD_ASYNC_TO_LDS_B128 (ASYNCcnt) so HBM/L2 latency
// hides under the 8 WMMAs per wave.  All LDS accesses use explicit AS3
// pointers so fragment reads lower to ds_load_b128.
// ---------------------------------------------------------------------------

typedef __bf16 bf16;
typedef __attribute__((ext_vector_type(16))) __bf16 v16bf;
typedef __attribute__((ext_vector_type(8)))  __bf16 v8bf;   // 16 bytes
typedef __attribute__((ext_vector_type(8)))  float  v8f;
typedef __attribute__((ext_vector_type(4)))  int    v4i;    // 16 bytes

#define AS1 __attribute__((address_space(1)))
#define AS3 __attribute__((address_space(3)))

#define BB   4
#define TT   2048
#define DD   1024
#define BT   (BB * TT)          // 8192 rows
#define LDSS 40                 // bf16 elems per LDS row: 32 + 8 pad (80B, 16B-aligned)
#define TILE (128 * LDSS)       // one LDS tile in bf16 elements

#if __has_builtin(__builtin_amdgcn_global_load_async_to_lds_b128)
#define GEMM_ASYNC 1
#else
#define GEMM_ASYNC 0
#endif

enum {
    EPI_BF16_BIAS      = 0,   // C = A*B + bias           -> bf16   (QKV)
    EPI_F32_SCALE_MASK = 1,   // C = A*B*scale, causal    -> f32    (scores)
    EPI_BF16           = 2,   // C = A*B                  -> bf16   (P@V)
    EPI_F32_BIAS       = 3,   // C = A*B + bias           -> f32    (proj, final)
    EPI_BF16_BIAS_GELU = 4,   // C = gelu(A*B + bias)     -> bf16   (MLP1)
};

__device__ __forceinline__ float gelu_tanh(float x) {
    float x3 = x * x * x;
    return 0.5f * x * (1.0f + tanhf(0.7978845608028654f * (x + 0.044715f * x3)));
}

__device__ __forceinline__ void wait_async0() {
#if __has_builtin(__builtin_amdgcn_s_wait_asynccnt)
    __builtin_amdgcn_s_wait_asynccnt(0);
#elif GEMM_ASYNC
    asm volatile("s_wait_asynccnt 0x0" ::: "memory");
#endif
}

// ---------------------------------------------------------------------------
// Generic WMMA GEMM: C[M,N] = A[M,K] (row-major bf16) * Bt[N,K]^T (row-major
// bf16, i.e. Bt[n][k] = B[k][n]).  M%128==0, N%128==0, K%32==0 (true for all
// shapes here).  Block = 256 threads = 8 waves; wave grid 2x4; each wave owns
// a 64x32 output region = 4x2 WMMA 16x16 tiles.
// ---------------------------------------------------------------------------
template <int EPI>
__global__ __launch_bounds__(256)
void gemm_wmma(const bf16* __restrict__ A, int lda, long sA,
               const bf16* __restrict__ Bt, int ldb, long sB,
               void* __restrict__ Cout, int ldc, long sC,
               const float* __restrict__ bias,
               int M, int N, int K, float scale)
{
    __shared__ bf16 smem[4 * TILE];        // [A0 | B0 | A1 | B1], ~41 KB
    AS3 bf16* sbase = (AS3 bf16*)smem;     // SSA-level cast only (no static init)

    const int bz = blockIdx.z;
    A  += (long)bz * sA;
    Bt += (long)bz * sB;

    const int m0 = blockIdx.y * 128;
    const int n0 = blockIdx.x * 128;

    const int tid  = threadIdx.x;
    const int lane = tid & 31;
    const int wave = tid >> 5;        // 0..7 (wave32)
    const int wm   = wave & 1;        // wave row   (2 x 64 rows)
    const int wn   = wave >> 1;       // wave col   (4 x 32 cols)

    const int lg = lane >> 4;         // lane group (0 or 1)
    const int lm = lane & 15;

    // global->LDS staging coords: thread t covers row tid>>1, 16-elem half tid&1
    const int lr = tid >> 1;          // 0..127
    const int lh = tid & 1;
    const long aRow = (long)(m0 + lr) * lda + 16 * lh;
    const long bRow = (long)(n0 + lr) * ldb + 16 * lh;
    const int  ldsOff = lr * LDSS + 16 * lh;    // elements

    v8f acc[4][2];
#pragma unroll
    for (int i = 0; i < 4; i++)
#pragma unroll
        for (int j = 0; j < 2; j++)
#pragma unroll
            for (int r = 0; r < 8; r++) acc[i][j][r] = 0.0f;

    union Frag { v16bf v; v8bf h[2]; };

#if GEMM_ASYNC
    // ---- async pipeline: DMA next tile into LDS while WMMAs run ----------
    auto issue_tile = [&](int k0, int pb) {
        const bf16* ga = A  + aRow + k0;
        const bf16* gb = Bt + bRow + k0;
        const int bsel = pb * (2 * TILE);
        AS3 bf16* la = sbase + bsel + ldsOff;
        AS3 bf16* lb = sbase + TILE + bsel + ldsOff;
        // imm offset applies to both global and LDS addresses (ISA 10.x)
        __builtin_amdgcn_global_load_async_to_lds_b128((AS1 v4i*)ga, (AS3 v4i*)la, 0,  0);
        __builtin_amdgcn_global_load_async_to_lds_b128((AS1 v4i*)ga, (AS3 v4i*)la, 16, 0);
        __builtin_amdgcn_global_load_async_to_lds_b128((AS1 v4i*)gb, (AS3 v4i*)lb, 0,  0);
        __builtin_amdgcn_global_load_async_to_lds_b128((AS1 v4i*)gb, (AS3 v4i*)lb, 16, 0);
    };
    issue_tile(0, 0);
#else
    // ---- register-prefetch pipeline --------------------------------------
    v8bf pa0, pa1, pb0, pb1;
    {
        const bf16* ga = A  + aRow;
        const bf16* gb = Bt + bRow;
        pa0 = *(const v8bf*)(ga);
        pa1 = *(const v8bf*)(ga + 8);
        pb0 = *(const v8bf*)(gb);
        pb1 = *(const v8bf*)(gb + 8);
    }
#endif

    int p = 0;
    for (int k0 = 0; k0 < K; k0 += 32) {
        const bool hasNext = (k0 + 32 < K);
        const int  bsel    = p * (2 * TILE);

#if GEMM_ASYNC
        wait_async0();          // tile p fully landed in LDS (this wave's part)
        __syncthreads();        // all waves' parts visible
        if (hasNext) issue_tile(k0 + 32, p ^ 1);   // overlaps with compute below
#else
        // stage prefetched regs into buffer p (readers of p finished >= 1 barrier ago)
        *(AS3 v8bf*)(sbase + bsel + ldsOff)            = pa0;
        *(AS3 v8bf*)(sbase + bsel + ldsOff + 8)        = pa1;
        *(AS3 v8bf*)(sbase + TILE + bsel + ldsOff)     = pb0;
        *(AS3 v8bf*)(sbase + TILE + bsel + ldsOff + 8) = pb1;
        if (hasNext) {          // issue next global loads before the barrier
            const bf16* ga = A  + aRow + k0 + 32;
            const bf16* gb = Bt + bRow + k0 + 32;
            pa0 = *(const v8bf*)(ga);
            pa1 = *(const v8bf*)(ga + 8);
            pb0 = *(const v8bf*)(gb);
            pb1 = *(const v8bf*)(gb + 8);
        }
        __syncthreads();        // tile p visible to all waves
#endif

        const AS3 bf16* As = sbase + bsel;
        const AS3 bf16* Bs = sbase + TILE + bsel;

        // A frags: lanes 0-15 -> row m, elems 0..7 = K 8g..8g+7, 8..15 = K 16+8g..
        Frag af[4];
#pragma unroll
        for (int i = 0; i < 4; i++) {
            const int row = wm * 64 + i * 16 + lm;
            af[i].h[0] = *(const AS3 v8bf*)(As + row * LDSS + 8 * lg);
            af[i].h[1] = *(const AS3 v8bf*)(As + row * LDSS + 16 + 8 * lg);
        }
        // B frags: lanes 0-15 -> column n = lm, elems 0..7 = K 8g..8g+7, 8..15 = K 16+8g..
        Frag bfr[2];
#pragma unroll
        for (int j = 0; j < 2; j++) {
            const int row = wn * 32 + j * 16 + lm;
            bfr[j].h[0] = *(const AS3 v8bf*)(Bs + row * LDSS + 8 * lg);
            bfr[j].h[1] = *(const AS3 v8bf*)(Bs + row * LDSS + 16 + 8 * lg);
        }

#pragma unroll
        for (int i = 0; i < 4; i++)
#pragma unroll
            for (int j = 0; j < 2; j++)
                acc[i][j] = __builtin_amdgcn_wmma_f32_16x16x32_bf16(
                    false, af[i].v, false, bfr[j].v,
                    (short)0, acc[i][j], false, false);

        p ^= 1;
    }

    // Epilogue.  C frag layout: lane l -> n = l&15, VGPR r -> m = r + 8*(l>>4).
    float* Cf = (float*)Cout;
    bf16*  Cb = (bf16*)Cout;
    const long cb = (long)bz * sC;

#pragma unroll
    for (int i = 0; i < 4; i++) {
        const int mBase = m0 + wm * 64 + i * 16 + 8 * lg;
#pragma unroll
        for (int j = 0; j < 2; j++) {
            const int nG = n0 + wn * 32 + j * 16 + lm;
            float bv = 0.0f;
            if (EPI == EPI_BF16_BIAS || EPI == EPI_F32_BIAS || EPI == EPI_BF16_BIAS_GELU)
                bv = bias[nG];
#pragma unroll
            for (int r = 0; r < 8; r++) {
                const int mG = mBase + r;
                float v = acc[i][j][r] + bv;
                if (EPI == EPI_F32_SCALE_MASK) {
                    v = acc[i][j][r] * scale;
                    if (nG > mG) v = -1e30f;       // causal mask
                }
                if (EPI == EPI_BF16_BIAS_GELU) v = gelu_tanh(v);
                if (EPI == EPI_F32_SCALE_MASK || EPI == EPI_F32_BIAS)
                    Cf[cb + (long)mG * ldc + nG] = v;
                else
                    Cb[cb + (long)mG * ldc + nG] = (bf16)v;
            }
        }
    }
}

// ---------------------------------------------------------------------------
// LayerNorm over rows of length D, cast to bf16.  One 256-thread block / row.
// ---------------------------------------------------------------------------
__global__ __launch_bounds__(256)
void layernorm_cast(const float* __restrict__ x, const float* __restrict__ g,
                    const float* __restrict__ be, bf16* __restrict__ out)
{
    const long row = blockIdx.x;
    const float* xr = x + row * DD;
    bf16* orow = out + row * DD;
    __shared__ float red[256];
    const int tid = threadIdx.x;

    float s = 0.0f;
    for (int i = tid; i < DD; i += 256) s += xr[i];
    red[tid] = s; __syncthreads();
    for (int off = 128; off > 0; off >>= 1) {
        if (tid < off) red[tid] += red[tid + off];
        __syncthreads();
    }
    const float mu = red[0] * (1.0f / DD);
    __syncthreads();

    float v = 0.0f;
    for (int i = tid; i < DD; i += 256) { float d = xr[i] - mu; v += d * d; }
    red[tid] = v; __syncthreads();
    for (int off = 128; off > 0; off >>= 1) {
        if (tid < off) red[tid] += red[tid + off];
        __syncthreads();
    }
    const float rstd = rsqrtf(red[0] * (1.0f / DD) + 1e-5f);

    for (int i = tid; i < DD; i += 256)
        orow[i] = (bf16)((xr[i] - mu) * rstd * g[i] + be[i]);
}

// ---------------------------------------------------------------------------
// Row softmax over length T, f32 in -> bf16 out.  One block / row.
// ---------------------------------------------------------------------------
__global__ __launch_bounds__(256)
void softmax_rows(const float* __restrict__ S, bf16* __restrict__ P)
{
    const long row = blockIdx.x;
    const float* sr = S + row * TT;
    bf16* pr = P + row * TT;
    __shared__ float red[256];
    const int tid = threadIdx.x;

    float mx = -3.0e38f;
    for (int i = tid; i < TT; i += 256) mx = fmaxf(mx, sr[i]);
    red[tid] = mx; __syncthreads();
    for (int off = 128; off > 0; off >>= 1) {
        if (tid < off) red[tid] = fmaxf(red[tid], red[tid + off]);
        __syncthreads();
    }
    const float m = red[0];
    __syncthreads();

    float sum = 0.0f;
    for (int i = tid; i < TT; i += 256) sum += expf(sr[i] - m);
    red[tid] = sum; __syncthreads();
    for (int off = 128; off > 0; off >>= 1) {
        if (tid < off) red[tid] += red[tid + off];
        __syncthreads();
    }
    const float inv = 1.0f / red[0];

    for (int i = tid; i < TT; i += 256)
        pr[i] = (bf16)(expf(sr[i] - m) * inv);
}

// ---------------------------------------------------------------------------
// Transpose-cast weight: Wt[n*K+k] = bf16(W[k*N+n]).  K,N multiples of 32.
// ---------------------------------------------------------------------------
__global__ __launch_bounds__(256)
void transpose_cast(const float* __restrict__ W, bf16* __restrict__ Wt,
                    int K, int N)
{
    __shared__ float tile[32][33];
    const int kb = blockIdx.y * 32, nb = blockIdx.x * 32;
    const int tx = threadIdx.x & 31, ty = threadIdx.x >> 5;   // ty 0..7
#pragma unroll
    for (int i = 0; i < 32; i += 8)
        tile[ty + i][tx] = W[(long)(kb + ty + i) * N + nb + tx];
    __syncthreads();
#pragma unroll
    for (int i = 0; i < 32; i += 8)
        Wt[(long)(nb + ty + i) * K + kb + tx] = (bf16)tile[tx][ty + i];
}

// ---------------------------------------------------------------------------
// Transpose V (bf16) out of the packed kqv buffer:
//   vt[b][d][t] = kqv[(b*T+t)*3D + 2D + d]
// ---------------------------------------------------------------------------
__global__ __launch_bounds__(256)
void transpose_v(const bf16* __restrict__ kqv, bf16* __restrict__ vt)
{
    __shared__ bf16 tile[32][33];
    const int b  = blockIdx.z;
    const int tb = blockIdx.y * 32, db = blockIdx.x * 32;
    const int tx = threadIdx.x & 31, ty = threadIdx.x >> 5;
#pragma unroll
    for (int i = 0; i < 32; i += 8)
        tile[ty + i][tx] = kqv[((long)b * TT + tb + ty + i) * (3 * DD) + 2 * DD + db + tx];
    __syncthreads();
#pragma unroll
    for (int i = 0; i < 32; i += 8)
        vt[(long)b * DD * TT + (long)(db + ty + i) * TT + tb + tx] = tile[tx][ty + i];
}

// ---------------------------------------------------------------------------
extern "C" void kernel_launch(void* const* d_in, const int* in_sizes, int n_in,
                              void* d_out, int out_size, void* d_ws, size_t ws_size,
                              hipStream_t stream)
{
    (void)in_sizes; (void)n_in; (void)out_size; (void)ws_size;

    const float* x     = (const float*)d_in[0];
    const float* ln1_g = (const float*)d_in[1];
    const float* ln1_b = (const float*)d_in[2];
    const float* Wqkv  = (const float*)d_in[3];
    const float* bqkv  = (const float*)d_in[4];
    const float* Wproj = (const float*)d_in[5];
    const float* bproj = (const float*)d_in[6];
    const float* ln2_g = (const float*)d_in[7];
    const float* ln2_b = (const float*)d_in[8];
    const float* W1    = (const float*)d_in[9];
    const float* b1    = (const float*)d_in[10];
    const float* W2    = (const float*)d_in[11];
    const float* b2    = (const float*)d_in[12];
    float* out = (float*)d_out;

    // ---- workspace carve-up (sizes all 256B-aligned by construction) ----
    char* ws = (char*)d_ws;
    size_t off = 0;
    auto carve = [&](size_t bytes) -> char* {
        char* p = ws + off;
        off += (bytes + 255) & ~(size_t)255;
        return p;
    };
    bf16* h      = (bf16*)carve((size_t)BT * DD * 2);            // 16 MB  (LN1 out)
    bf16* wqkvt  = (bf16*)carve((size_t)3 * DD * DD * 2);        //  6 MB
    bf16* wprojt = (bf16*)carve((size_t)DD * DD * 2);            //  2 MB
    bf16* w1t    = (bf16*)carve((size_t)4 * DD * DD * 2);        //  8 MB
    bf16* w2t    = (bf16*)carve((size_t)4 * DD * DD * 2);        //  8 MB
    bf16* kqv    = (bf16*)carve((size_t)BT * 3 * DD * 2);        // 48 MB
    bf16* vt     = (bf16*)carve((size_t)BB * DD * TT * 2);       // 16 MB
    char* sc_raw =        carve((size_t)BB * TT * TT * 4);       // 64 MB  (scores f32)
    bf16* p      = (bf16*)carve((size_t)BB * TT * TT * 2);       // 32 MB  (softmax out)
    bf16* o      = (bf16*)carve((size_t)BT * DD * 2);            // 16 MB  (attn out)
    float* scores   = (float*)sc_raw;
    bf16*  m        = (bf16*)sc_raw;     // MLP hidden reuses dead scores (64 MB)
    float* proj_f32 = (float*)kqv;       // proj out reuses dead kqv (32 <= 48 MB)
    bf16*  h2       = h;                 // LN2 out reuses dead h

    const dim3 blk(256);
    const bf16* kptr = kqv;              // split order: k, q, v
    const bf16* qptr = kqv + DD;
    const float scale = 0.022097086912079608f;   // 1/sqrt(T), T = 2048

    // 1) weight transpose-casts to bf16
    transpose_cast<<<dim3(3 * DD / 32, DD / 32), blk, 0, stream>>>(Wqkv, wqkvt, DD, 3 * DD);
    transpose_cast<<<dim3(DD / 32, DD / 32),     blk, 0, stream>>>(Wproj, wprojt, DD, DD);
    transpose_cast<<<dim3(4 * DD / 32, DD / 32), blk, 0, stream>>>(W1, w1t, DD, 4 * DD);
    transpose_cast<<<dim3(DD / 32, 4 * DD / 32), blk, 0, stream>>>(W2, w2t, 4 * DD, DD);

    // 2) LN1: x -> h (bf16)
    layernorm_cast<<<dim3(BT), blk, 0, stream>>>(x, ln1_g, ln1_b, h);

    // 3) QKV GEMM: kqv = h @ Wqkv + bqkv   [8192 x 3072]
    gemm_wmma<EPI_BF16_BIAS><<<dim3(3 * DD / 128, BT / 128, 1), blk, 0, stream>>>(
        h, DD, 0, wqkvt, DD, 0, kqv, 3 * DD, 0, bqkv, BT, 3 * DD, DD, 1.0f);

    // 4) V transpose: vt[b][d][t]
    transpose_v<<<dim3(DD / 32, TT / 32, BB), blk, 0, stream>>>(kqv, vt);

    // 5) scores = q @ k^T * (1/sqrt(T)), causal mask   [4 x 2048 x 2048] f32
    gemm_wmma<EPI_F32_SCALE_MASK><<<dim3(TT / 128, TT / 128, BB), blk, 0, stream>>>(
        qptr, 3 * DD, (long)TT * 3 * DD,
        kptr, 3 * DD, (long)TT * 3 * DD,
        scores, TT, (long)TT * TT, nullptr, TT, TT, DD, scale);

    // 6) softmax rows -> p (bf16)
    softmax_rows<<<dim3(BB * TT), blk, 0, stream>>>(scores, p);

    // 7) o = p @ v   [4 x 2048 x 1024] -> bf16 (contiguous as [8192 x 1024])
    gemm_wmma<EPI_BF16><<<dim3(DD / 128, TT / 128, BB), blk, 0, stream>>>(
        p, TT, (long)TT * TT,
        vt, TT, (long)DD * TT,
        o, DD, (long)TT * DD, nullptr, TT, DD, TT, 1.0f);

    // 8) proj: proj_f32 = o @ Wproj + bproj   [8192 x 1024] f32
    gemm_wmma<EPI_F32_BIAS><<<dim3(DD / 128, BT / 128, 1), blk, 0, stream>>>(
        o, DD, 0, wprojt, DD, 0, proj_f32, DD, 0, bproj, BT, DD, DD, 1.0f);

    // 9) LN2: proj_f32 -> h2 (bf16)
    layernorm_cast<<<dim3(BT), blk, 0, stream>>>(proj_f32, ln2_g, ln2_b, h2);

    // 10) MLP1: m = gelu(h2 @ W1 + b1)   [8192 x 4096] bf16
    gemm_wmma<EPI_BF16_BIAS_GELU><<<dim3(4 * DD / 128, BT / 128, 1), blk, 0, stream>>>(
        h2, DD, 0, w1t, DD, 0, m, 4 * DD, 0, b1, BT, 4 * DD, DD, 1.0f);

    // 11) MLP2: out = m @ W2 + b2   [8192 x 1024] f32 -> d_out
    gemm_wmma<EPI_F32_BIAS><<<dim3(DD / 128, BT / 128, 1), blk, 0, stream>>>(
        m, 4 * DD, 0, w2t, 4 * DD, 0, out, DD, 0, b2, BT, DD, 4 * DD, 1.0f);
}
